// GCNSynthetic_111669150054
// MI455X (gfx1250) — matrix operand order repaired and checked by reading, not compiled, not run on hardware
//
#include <hip/hip_runtime.h>
#include <stdint.h>

typedef uint16_t u16;
typedef uint32_t u32;
typedef __attribute__((ext_vector_type(16))) __bf16 bf16x16;
typedef __attribute__((ext_vector_type(8)))  float   f32x8;
typedef __attribute__((ext_vector_type(4)))  u32     u32x4;

union FragU {
    u32     w[8];
    u32x4   q[2];
    bf16x16 v;
};

// f32 pair -> packed bf16 (round-half-up): bump bit15, then splice the two
// high halves with one v_perm_b32.  3 VALU ops per pair.
__device__ __forceinline__ u32 pack2_bf16(float lo, float hi) {
    u32 a = __float_as_uint(lo) + 0x8000u;
    u32 b = __float_as_uint(hi) + 0x8000u;
    return __builtin_amdgcn_perm(b, a, 0x07060302u);  // {b[31:16], a[31:16]}
}

// ---------------------------------------------------------------------------
// Weight prep: Wt[n*K + k] = bf16(W[k*Ncol + n])   (column-major bf16 copy)
// ---------------------------------------------------------------------------
__global__ void conv_transpose_bf16(const float* __restrict__ W,
                                    u16* __restrict__ Wt, int K, int Ncol) {
    int idx = blockIdx.x * 256 + threadIdx.x;
    if (idx >= K * Ncol) return;
    int n = idx / K;
    int k = idx - n * K;
    u32 a = __float_as_uint(W[(size_t)k * Ncol + n]);
    Wt[idx] = (u16)((a + 0x7FFFu + ((a >> 16) & 1u)) >> 16);
}

// ---------------------------------------------------------------------------
// C[M,256] = A[M,K] @ B[K,256], B transposed bf16 (Bt[n*K+k]).
// Block = 256 threads = 8 waves arranged 2(M) x 4(N) over a 128x256 tile.
// Each wave: 4 M-subtiles x 4 N-subtiles -> every fragment feeds 4 WMMAs.
// ---------------------------------------------------------------------------
__global__ __launch_bounds__(256)
void gemm_bf16_wmma(const float* __restrict__ A, const u16* __restrict__ Bt,
                    float* __restrict__ C, int M, int K) {
    const int lane  = threadIdx.x & 31;
    const int wave  = threadIdx.x >> 5;
    const int waveM = wave >> 2;                 // 0..1
    const int waveN = wave & 3;                  // 0..3
    const int mBase = blockIdx.x * 128 + waveM * 64;
    if (mBase >= M) return;                      // wave-uniform exit

    const int lh  = lane >> 4;
    const int l15 = lane & 15;

    const float* aRow[4];
    #pragma unroll
    for (int i = 0; i < 4; ++i) {
        int r = mBase + i * 16 + l15;
        if (r >= M) r = M - 1;                   // clamp; stores guarded
        aRow[i] = A + (size_t)r * K + lh * 8;
    }
    const u16* bRow[4];
    #pragma unroll
    for (int j = 0; j < 4; ++j)
        bRow[j] = Bt + (size_t)(waveN * 64 + j * 16 + l15) * K + lh * 8;

    f32x8 acc[4][4] = {};

    for (int k0 = 0; k0 < K; k0 += 32) {
        // ---- issue all loads for this k-step up front ----
        float4 araw[4][4];
        #pragma unroll
        for (int i = 0; i < 4; ++i) {
            const float* ap = aRow[i] + k0;
            araw[i][0] = *(const float4*)(ap);
            araw[i][1] = *(const float4*)(ap + 4);
            araw[i][2] = *(const float4*)(ap + 16);
            araw[i][3] = *(const float4*)(ap + 20);
        }
        FragU bf[4];
        #pragma unroll
        for (int j = 0; j < 4; ++j) {
            const u16* bp = bRow[j] + k0;
            bf[j].q[0] = *(const u32x4*)(bp);
            bf[j].q[1] = *(const u32x4*)(bp + 16);
        }
        // ---- convert A to bf16 fragments (short add+perm chains) ----
        FragU af[4];
        #pragma unroll
        for (int i = 0; i < 4; ++i) {
            #pragma unroll
            for (int p = 0; p < 4; ++p) {
                af[i].w[2 * p]     = pack2_bf16(araw[i][p].x, araw[i][p].y);
                af[i].w[2 * p + 1] = pack2_bf16(araw[i][p].z, araw[i][p].w);
            }
        }
        // ---- 16 WMMAs, each fragment reused 4x ----
        #pragma unroll
        for (int i = 0; i < 4; ++i)
            #pragma unroll
            for (int j = 0; j < 4; ++j)
                acc[i][j] = __builtin_amdgcn_wmma_f32_16x16x32_bf16(
                    false, af[i].v, false, bf[j].v, (short)0, acc[i][j],
                    false, false);
    }

    #pragma unroll
    for (int i = 0; i < 4; ++i)
        #pragma unroll
        for (int j = 0; j < 4; ++j)
            #pragma unroll
            for (int t = 0; t < 8; ++t) {
                int r = mBase + i * 16 + t + lh * 8;
                if (r < M)
                    C[(size_t)r * 256 + waveN * 64 + j * 16 + l15] = acc[i][j][t];
            }
}

// ---------------------------------------------------------------------------
// CSR build: zero -> histogram -> exclusive scan -> scatter fill
// ---------------------------------------------------------------------------
__global__ void zero_i32(int* __restrict__ p, int n) {
    int i = blockIdx.x * 256 + threadIdx.x;
    if (i < n) p[i] = 0;
}

__global__ void hist_kernel(const int* __restrict__ rows, int* __restrict__ cnt,
                            int E) {
    int e = blockIdx.x * 256 + threadIdx.x;
    if (e < E) atomicAdd(&cnt[rows[e]], 1);
}

// single-block 1024-thread exclusive scan over N counters
__global__ __launch_bounds__(1024)
void excl_scan(const int* __restrict__ cnt, int* __restrict__ offs,
               int* __restrict__ cursor, int N) {
    __shared__ int sd[1024];
    __shared__ int running;
    const int t = threadIdx.x;
    if (t == 0) running = 0;
    __syncthreads();
    for (int base = 0; base < N; base += 1024) {
        int i = base + t;
        int v = (i < N) ? cnt[i] : 0;
        sd[t] = v;
        __syncthreads();
        #pragma unroll
        for (int off = 1; off < 1024; off <<= 1) {
            int y = (t >= off) ? sd[t - off] : 0;
            __syncthreads();
            sd[t] += y;
            __syncthreads();
        }
        int excl = sd[t] - v + running;
        if (i < N) { offs[i] = excl; cursor[i] = excl; }
        int tot = sd[1023];
        __syncthreads();
        if (t == 0) running += tot;
        __syncthreads();
    }
    if (t == 0) offs[N] = running;
}

// pack (col, val) into one 8-byte record per edge
__global__ void fill_kernel(const int* __restrict__ rows,
                            const int* __restrict__ cols,
                            const float* __restrict__ vals,
                            int* __restrict__ cursor, int2* __restrict__ edges,
                            int E) {
    int e = blockIdx.x * 256 + threadIdx.x;
    if (e >= E) return;
    int pos = atomicAdd(&cursor[rows[e]], 1);
    int2 rec;
    rec.x = cols[e];
    rec.y = __float_as_int(vals[e]);
    edges[pos] = rec;
}

// ---------------------------------------------------------------------------
// Atomic-free row-parallel SpMM with fused bias (+ optional ReLU):
// out[r,:] = act( b + sum_e val[e] * S[col[e],:] ).  One wave per row,
// 8 f32 accumulators per lane; next edge record prefetched each iteration.
// ---------------------------------------------------------------------------
__global__ __launch_bounds__(256)
void spmm_csr(const int* __restrict__ offs, const int2* __restrict__ edges,
              const float* __restrict__ S, const float* __restrict__ bias,
              float* __restrict__ out, int N, int doRelu) {
    int r = blockIdx.x * 8 + (threadIdx.x >> 5);
    if (r >= N) return;
    int lane = threadIdx.x & 31;

    float4 a0 = *(const float4*)(bias + lane * 8);
    float4 a1 = *(const float4*)(bias + lane * 8 + 4);

    int beg = offs[r], end = offs[r + 1];
    if (beg < end) {
        int2 ed = edges[beg];
        for (int e = beg; e < end; ++e) {
            int   c = ed.x;
            float v = __int_as_float(ed.y);
            const float* src = S + (size_t)c * 256 + lane * 8;
            float4 s0 = *(const float4*)(src);
            float4 s1 = *(const float4*)(src + 4);
            if (e + 1 < end) ed = edges[e + 1];   // prefetch next record
            a0.x = fmaf(v, s0.x, a0.x); a0.y = fmaf(v, s0.y, a0.y);
            a0.z = fmaf(v, s0.z, a0.z); a0.w = fmaf(v, s0.w, a0.w);
            a1.x = fmaf(v, s1.x, a1.x); a1.y = fmaf(v, s1.y, a1.y);
            a1.z = fmaf(v, s1.z, a1.z); a1.w = fmaf(v, s1.w, a1.w);
        }
    }
    if (doRelu) {
        a0.x = fmaxf(a0.x, 0.f); a0.y = fmaxf(a0.y, 0.f);
        a0.z = fmaxf(a0.z, 0.f); a0.w = fmaxf(a0.w, 0.f);
        a1.x = fmaxf(a1.x, 0.f); a1.y = fmaxf(a1.y, 0.f);
        a1.z = fmaxf(a1.z, 0.f); a1.w = fmaxf(a1.w, 0.f);
    }
    float* dst = out + (size_t)r * 256 + lane * 8;
    *(float4*)(dst)     = a0;
    *(float4*)(dst + 4) = a1;
}

// ---------------------------------------------------------------------------
// logits = concat(x1,x2,x3) @ linW^T + linb ; out = log_softmax(logits)
// ---------------------------------------------------------------------------
__global__ __launch_bounds__(256)
void head_kernel(const float* __restrict__ x1, const float* __restrict__ x2,
                 const float* __restrict__ x3, const float* __restrict__ linW,
                 const float* __restrict__ linb, float* __restrict__ out, int N) {
    int n = blockIdx.x * 8 + (threadIdx.x >> 5);
    if (n >= N) return;
    int lane = threadIdx.x & 31;

    const float* p1 = x1 + (size_t)n * 256;
    const float* p2 = x2 + (size_t)n * 256;
    const float* p3 = x3 + (size_t)n * 256;

    float acc[8] = {};
    for (int k = lane; k < 256; k += 32) {
        float v1 = p1[k], v2 = p2[k], v3 = p3[k];
        #pragma unroll
        for (int c = 0; c < 8; ++c) {
            const float* w = linW + c * 768;
            acc[c] += v1 * w[k] + v2 * w[256 + k] + v3 * w[512 + k];
        }
    }
    #pragma unroll
    for (int c = 0; c < 8; ++c) {
        #pragma unroll
        for (int off = 16; off > 0; off >>= 1)
            acc[c] += __shfl_xor(acc[c], off, 32);
    }
    if (lane == 0) {
        float logit[8];
        float m = -3.4e38f;
        #pragma unroll
        for (int c = 0; c < 8; ++c) { logit[c] = acc[c] + linb[c]; m = fmaxf(m, logit[c]); }
        float s = 0.0f;
        #pragma unroll
        for (int c = 0; c < 8; ++c) s += expf(logit[c] - m);
        float lse = m + logf(s);
        #pragma unroll
        for (int c = 0; c < 8; ++c) out[(size_t)n * 8 + c] = logit[c] - lse;
    }
}

// ---------------------------------------------------------------------------
extern "C" void kernel_launch(void* const* d_in, const int* in_sizes, int n_in,
                              void* d_out, int out_size, void* d_ws, size_t ws_size,
                              hipStream_t stream) {
    const float* x       = (const float*)d_in[0];
    const int*   adj_row = (const int*)d_in[1];
    const int*   adj_col = (const int*)d_in[2];
    const float* adj_val = (const float*)d_in[3];
    const float* W1      = (const float*)d_in[4];
    const float* b1      = (const float*)d_in[5];
    const float* W2      = (const float*)d_in[6];
    const float* b2      = (const float*)d_in[7];
    const float* W3      = (const float*)d_in[8];
    const float* b3      = (const float*)d_in[9];
    const float* linW    = (const float*)d_in[10];
    const float* linb    = (const float*)d_in[11];

    const int NFEAT = 512, NHID = 256;
    const int N = in_sizes[0] / NFEAT;
    const int E = in_sizes[1];
    const size_t NH = (size_t)N * NHID;

    // workspace: S | x1 | x2 | x3 | W1t W2t W3t | cnt offs cursor | edges
    float* S  = (float*)d_ws;
    float* x1 = S  + NH;
    float* x2 = x1 + NH;
    float* x3 = x2 + NH;
    u16* W1t = (u16*)(x3 + NH);
    u16* W2t = W1t + (size_t)NFEAT * NHID;
    u16* W3t = W2t + (size_t)NHID * NHID;
    int* cnt    = (int*)(W3t + (size_t)NHID * NHID);
    int* offs   = cnt + N;
    int* cursor = offs + N + 1;
    int2* edges = (int2*)(cursor + N + 1);   // +1 keeps 8B alignment

    const int eb256 = (E + 255) / 256;
    const int nb256 = (N + 255) / 256;
    const int gb    = (N + 127) / 128;
    const int rb    = (N + 7) / 8;

    // ---- CSR build (once, reused by all 3 layers) ----
    zero_i32<<<nb256, 256, 0, stream>>>(cnt, N);
    hist_kernel<<<eb256, 256, 0, stream>>>(adj_row, cnt, E);
    excl_scan<<<1, 1024, 0, stream>>>(cnt, offs, cursor, N);
    fill_kernel<<<eb256, 256, 0, stream>>>(adj_row, adj_col, adj_val,
                                           cursor, edges, E);

    // ---- weight prep ----
    conv_transpose_bf16<<<(NFEAT * NHID + 255) / 256, 256, 0, stream>>>(W1, W1t, NFEAT, NHID);
    conv_transpose_bf16<<<(NHID * NHID + 255) / 256, 256, 0, stream>>>(W2, W2t, NHID, NHID);
    conv_transpose_bf16<<<(NHID * NHID + 255) / 256, 256, 0, stream>>>(W3, W3t, NHID, NHID);

    // ---- layer 1: x1 = relu(spmm(x @ W1) + b1) ----
    gemm_bf16_wmma<<<gb, 256, 0, stream>>>(x, W1t, S, N, NFEAT);
    spmm_csr<<<rb, 256, 0, stream>>>(offs, edges, S, b1, x1, N, 1);

    // ---- layer 2 ----
    gemm_bf16_wmma<<<gb, 256, 0, stream>>>(x1, W2t, S, N, NHID);
    spmm_csr<<<rb, 256, 0, stream>>>(offs, edges, S, b2, x2, N, 1);

    // ---- layer 3 (no relu) ----
    gemm_bf16_wmma<<<gb, 256, 0, stream>>>(x2, W3t, S, N, NHID);
    spmm_csr<<<rb, 256, 0, stream>>>(offs, edges, S, b3, x3, N, 0);

    // ---- head ----
    head_kernel<<<rb, 256, 0, stream>>>(x1, x2, x3, linW, linb, (float*)d_out, N);
}
